// CommNetActor_70858370449763
// MI455X (gfx1250) — matrix operand order repaired and checked by reading, not compiled, not run on hardware
//
#include <hip/hip_runtime.h>
#include <hip/hip_bf16.h>

typedef __attribute__((ext_vector_type(16))) _Float16     v16h;
typedef __attribute__((ext_vector_type(8)))  _Float16     v8h;
typedef __attribute__((ext_vector_type(8)))  float        v8f;
typedef __attribute__((ext_vector_type(4)))  unsigned int u32x4;

#define AS    264      // activation row stride in halfs (256 data + 8 pad)
#define WAVES 8

// LDS weight segment offsets (halfs), padded stride K+8 per row
#define W_ENC 0        // 128 x 72
#define W_FC1 9216     // 128 x 136
#define W_FC2 26624    // 128 x 136
#define W_FC3 44032    // 128 x 136
#define W_CL4 61440    // 128 x 264
#define W_DEC 95232    // 16  x 520
#define W_TOT 103552   // halfs = 207104 B

union HV { u32x4 q[2]; v16h v; };

__device__ __forceinline__ u32x4 ld16(const _Float16* p) { return *(const u32x4*)p; }
__device__ __forceinline__ void  st16(_Float16* p, u32x4 x) { *(u32x4*)p = x; }

// ---------------------------------------------------------------------------
// One 16-row x 128-col layer for a single wave.  kb-outer / nt-inner with 8
// live accumulators -> 8 independent WMMAs per A fragment (pipelines LDS B
// loads under WMMA execution instead of s_wait_dscnt 0 before every WMMA).
//   act : wave's [16][AS] f16 staging (input cols [0,K), output cols [0,128))
//   wt  : LDS weights, transposed [128 rows=n][K+8 stride] f16
//   ACT : 0=none, 1=relu, 2=sigmoid
// ---------------------------------------------------------------------------
template<int K, int ACT>
__device__ __forceinline__ void run_layer(_Float16 (*act)[AS],
                                          const _Float16* wt,
                                          const float* __restrict__ bias,
                                          int l16, bool hi)
{
  constexpr int NKB = K / 32;
  constexpr int KP  = K + 8;

  v8f acc[8];
#pragma unroll
  for (int nt = 0; nt < 8; ++nt) acc[nt] = (v8f){0.f,0.f,0.f,0.f,0.f,0.f,0.f,0.f};

  const _Float16* arow = &act[l16][0];
  const int hoff = hi ? 8 : 0;   // A layout: lane<16 chunks {+0,+16}, lane>=16 {+8,+24}
  const int hk   = hi ? 8 : 0;   // B layout: lane<16 chunks {+0,+16}, lane>=16 {+8,+24}

#pragma unroll
  for (int kb = 0; kb < NKB; ++kb) {
    HV a;
    a.q[0] = ld16(arow + kb * 32 + hoff);
    a.q[1] = ld16(arow + kb * 32 + hoff + 16);
#pragma unroll
    for (int nt = 0; nt < 8; ++nt) {
      const _Float16* wrow = wt + (nt * 16 + l16) * KP + hk + kb * 32;
      HV b;
      b.q[0] = ld16(wrow);
      b.q[1] = ld16(wrow + 16);
      acc[nt] = __builtin_amdgcn_wmma_f32_16x16x32_f16(false, a.v, false, b.v,
                                                       (short)0, acc[nt], false, false);
    }
  }

  const int r0 = hi ? 8 : 0;     // C/D layout: VGPR v holds row v (+8 for hi lanes)
#pragma unroll
  for (int nt = 0; nt < 8; ++nt) {
    const int n = nt * 16 + l16;
    const float bv = bias[n];
#pragma unroll
    for (int v = 0; v < 8; ++v) {
      float x = acc[nt][v] + bv;
      if (ACT == 1) x = x > 0.f ? x : 0.f;
      if (ACT == 2) x = 1.f / (1.f + __expf(-x));
      act[r0 + v][n] = (_Float16)x;
    }
  }
}

// ---------------------------------------------------------------------------
// Fused CommNet actor forward. grid = 2048 blocks x 256 threads.
// Block = 32 batch entries = 128 (b,agent) rows; wave w owns rows
// [w*16, w*16+16) = batch entries [w*4, w*4+4).  ALL weights stay resident in
// LDS (~271 KB total of 320 KB/WGP), so there are no inter-layer barriers:
// waves run enc->cl4 completely independently; one barrier before dec.
// ---------------------------------------------------------------------------
__global__ __launch_bounds__(256, 1) void commnet_fused(
    const float* __restrict__ O,
    const float* __restrict__ enc_b, const float* __restrict__ fc1_b,
    const float* __restrict__ fc2_b, const float* __restrict__ fc3_b,
    const float* __restrict__ cl4_b, const float* __restrict__ dec_b,
    const _Float16* __restrict__ WT,
    float* __restrict__ out)
{
  __shared__ __align__(16) _Float16 s_wt[W_TOT];           // 207104 B, all layers
  __shared__ __align__(16) _Float16 s_act[WAVES][16][AS];  // 67584 B
  __shared__ float s_logits[32][17];

  const int  tid  = threadIdx.x;
  const int  wave = tid >> 5;
  const int  lane = tid & 31;
  const int  l16  = lane & 15;
  const bool hi   = lane >= 16;

  auto load_wt = [&](_Float16* dst, const _Float16* __restrict__ src, int rows, int K) {
    const int KP  = K + 8;
    const int cpr = K >> 3;            // 16B chunks per row
    const int tot = rows * cpr;
    for (int i = tid; i < tot; i += 256) {
      const int r = i / cpr;
      const int c = (i - r * cpr) << 3;
      st16(dst + r * KP + c, ld16(src + r * K + c));
    }
  };

  // ---- load ALL weights to LDS once + stage O tile (f32 -> f16, cols [0,64)) ----
  load_wt(s_wt + W_ENC, WT + 0,     128, 64);
  load_wt(s_wt + W_FC1, WT + 8192,  128, 128);
  load_wt(s_wt + W_FC2, WT + 24576, 128, 128);
  load_wt(s_wt + W_FC3, WT + 40960, 128, 128);
  load_wt(s_wt + W_CL4, WT + 57344, 256 == 256 ? 128 : 128, 256);
  load_wt(s_wt + W_DEC, WT + 90112, 16,  512);
  {
    const size_t row  = (size_t)blockIdx.x * 128 + wave * 16 + (lane >> 1);
    const float* orow = O + row * 64 + (lane & 1) * 32;
    _Float16*    drow = &s_act[wave][lane >> 1][(lane & 1) * 32];
#pragma unroll
    for (int j = 0; j < 32; j += 4) {
      float4 f = *(const float4*)(orow + j);
      drow[j + 0] = (_Float16)f.x; drow[j + 1] = (_Float16)f.y;
      drow[j + 2] = (_Float16)f.z; drow[j + 3] = (_Float16)f.w;
    }
  }
  __syncthreads();

  // ---- barrier-free layer pipeline (per-wave private tiles) ----
  run_layer<64,  2>(s_act[wave], s_wt + W_ENC, enc_b, l16, hi);  // sigmoid
  run_layer<128, 1>(s_act[wave], s_wt + W_FC1, fc1_b, l16, hi);  // relu
  run_layer<128, 1>(s_act[wave], s_wt + W_FC2, fc2_b, l16, hi);  // relu
  run_layer<128, 1>(s_act[wave], s_wt + W_FC3, fc3_b, l16, hi);  // relu -> H3

  // ---- comm: C[b,i] = (sum_j H3[b,j] - H3[b,i]) / 4, into cols [128,256) ----
  // 64 tasks = 16 column-chunks x 4 row-groups; 2 tasks/lane, 16B vector ops.
  {
    _Float16 (*h)[AS] = s_act[wave];
    const _Float16 q = (_Float16)0.25f;
#pragma unroll
    for (int t = 0; t < 2; ++t) {
      const int task = lane + t * 32;
      const int g  = task & 3;          // row group (batch entry in tile)
      const int c0 = (task >> 2) * 8;   // column chunk
      v8h h0 = *(const v8h*)&h[4 * g + 0][c0];
      v8h h1 = *(const v8h*)&h[4 * g + 1][c0];
      v8h h2 = *(const v8h*)&h[4 * g + 2][c0];
      v8h h3 = *(const v8h*)&h[4 * g + 3][c0];
      v8h s  = (h0 + h1) + (h2 + h3);
      *(v8h*)&h[4 * g + 0][128 + c0] = (s - h0) * q;
      *(v8h*)&h[4 * g + 1][128 + c0] = (s - h1) * q;
      *(v8h*)&h[4 * g + 2][128 + c0] = (s - h2) * q;
      *(v8h*)&h[4 * g + 3][128 + c0] = (s - h3) * q;
    }
  }

  run_layer<256, 0>(s_act[wave], s_wt + W_CL4, cl4_b, l16, hi);  // H4, no activation
  __syncthreads();   // dec reads other waves' H4 tiles

  // ---- dec: per batch entry bb, K=512 over agent-major concat of H4 rows ----
  if (wave < 2) {
    const int bb = wave * 16 + l16;                      // this lane's A row
    const _Float16* wrow = s_wt + W_DEC + l16 * 520 + (hi ? 8 : 0);
    v8f c = {0.f,0.f,0.f,0.f,0.f,0.f,0.f,0.f};
#pragma unroll
    for (int kb = 0; kb < 16; ++kb) {
      const int c0 = kb * 32 + (hi ? 8 : 0);
      const int c1 = c0 + 16;
      HV a;
      a.q[0] = ld16(&s_act[bb >> 2][(bb & 3) * 4 + (c0 >> 7)][c0 & 127]);
      a.q[1] = ld16(&s_act[bb >> 2][(bb & 3) * 4 + (c1 >> 7)][c1 & 127]);
      HV b;
      b.q[0] = ld16(wrow + kb * 32);
      b.q[1] = ld16(wrow + kb * 32 + 16);
      c = __builtin_amdgcn_wmma_f32_16x16x32_f16(false, a.v, false, b.v,
                                                 (short)0, c, false, false);
    }
    const float bv = dec_b[l16];
    const int r0 = wave * 16 + (hi ? 8 : 0);
#pragma unroll
    for (int v = 0; v < 8; ++v) s_logits[r0 + v][l16] = c[v] + bv;
  }
  __syncthreads();

  // ---- softmax: one thread per batch entry ----
  if (tid < 32) {
    float m = -3.4e38f;
#pragma unroll
    for (int j = 0; j < 16; ++j) m = fmaxf(m, s_logits[tid][j]);
    float e[16], s = 0.f;
#pragma unroll
    for (int j = 0; j < 16; ++j) { e[j] = __expf(s_logits[tid][j] - m); s += e[j]; }
    const float inv = 1.f / s;
    float* o = out + ((size_t)blockIdx.x * 32 + tid) * 16;
#pragma unroll
    for (int j = 0; j < 16; ++j) o[j] = e[j] * inv;
  }
}

// ---------------------------------------------------------------------------
// Prep: transpose + f32->f16 weights into d_ws. wt[n*din + k] = w[k*dout + n].
// ---------------------------------------------------------------------------
__global__ void wt_prep(const float* __restrict__ w, _Float16* __restrict__ wt,
                        int din, int dout)
{
  const int i = blockIdx.x * 256 + threadIdx.x;
  if (i < din * dout) {
    const int n = i / din;
    const int k = i - n * din;
    wt[i] = (_Float16)w[(size_t)k * dout + n];
  }
}

// ws layout (halfs): enc@0(8192) fc1@8192 fc2@24576 fc3@40960 cl4@57344 dec@90112
extern "C" void kernel_launch(void* const* d_in, const int* in_sizes, int n_in,
                              void* d_out, int out_size, void* d_ws, size_t ws_size,
                              hipStream_t stream) {
  const float* O     = (const float*)d_in[0];
  const float* enc_w = (const float*)d_in[1];
  const float* enc_b = (const float*)d_in[2];
  const float* fc1_w = (const float*)d_in[3];
  const float* fc1_b = (const float*)d_in[4];
  const float* fc2_w = (const float*)d_in[5];
  const float* fc2_b = (const float*)d_in[6];
  const float* fc3_w = (const float*)d_in[7];
  const float* fc3_b = (const float*)d_in[8];
  const float* cl4_w = (const float*)d_in[9];
  const float* cl4_b = (const float*)d_in[10];
  const float* dec_w = (const float*)d_in[11];
  const float* dec_b = (const float*)d_in[12];
  _Float16* WT = (_Float16*)d_ws;

  wt_prep<<<32,  256, 0, stream>>>(enc_w, WT + 0,     64,  128);
  wt_prep<<<64,  256, 0, stream>>>(fc1_w, WT + 8192,  128, 128);
  wt_prep<<<64,  256, 0, stream>>>(fc2_w, WT + 24576, 128, 128);
  wt_prep<<<64,  256, 0, stream>>>(fc3_w, WT + 40960, 128, 128);
  wt_prep<<<128, 256, 0, stream>>>(cl4_w, WT + 57344, 256, 128);
  wt_prep<<<32,  256, 0, stream>>>(dec_w, WT + 90112, 512, 16);

  commnet_fused<<<2048, 256, 0, stream>>>(O, enc_b, fc1_b, fc2_b, fc3_b,
                                          cl4_b, dec_b, WT, (float*)d_out);
}
